// DecoderLayer_21852793602874
// MI455X (gfx1250) — matrix operand order repaired and checked by reading, not compiled, not run on hardware
//
#include <hip/hip_runtime.h>
#include <hip/hip_bf16.h>

// ---------------------------------------------------------------------------
// MI455X (gfx1250) decoder layer: bf16 WMMA everywhere, f32 accumulate.
// GEMMs stage tiles with GLOBAL_LOAD_ASYNC_TO_LDS_B128 (ASYNCcnt), double-
// buffered LDS, one barrier per K-step.
// ---------------------------------------------------------------------------

typedef __attribute__((ext_vector_type(16))) __bf16 v16bf;
typedef __attribute__((ext_vector_type(8)))  __bf16 bf16x8;
typedef __attribute__((ext_vector_type(4)))  __bf16 bf16x4;
typedef __attribute__((ext_vector_type(8)))  float  v8f;
typedef __attribute__((ext_vector_type(4)))  float  f32x4;

constexpr int kDim   = 2048;
constexpr int kHd    = 128;
constexpr int kNH    = 16;
constexpr int kN     = 4;
constexpr int kT     = 2048;
constexpr int kRows  = kN * kT;            // 8192
constexpr float kEps = 1e-6f;

__device__ __forceinline__ v16bf mk16(bf16x8 lo, bf16x8 hi) {
  return __builtin_shufflevector(lo, hi, 0,1,2,3,4,5,6,7,8,9,10,11,12,13,14,15);
}

__device__ __forceinline__ v8f wmma_bf16(v16bf a, v16bf b, v8f c) {
  // v_wmma_f32_16x16x32_bf16
  return __builtin_amdgcn_wmma_f32_16x16x32_bf16(
      /*neg_a=*/false, a, /*neg_b=*/false, b,
      /*c_mod=*/(short)0, c, /*reuse_a=*/false, /*reuse_b=*/false);
}

// Wave-relative LDS byte offset of a generic pointer to a __shared__ object.
__device__ __forceinline__ unsigned lds_off(const void* p) {
  return (unsigned)(unsigned long long)(const __attribute__((address_space(3))) char*)p;
}

// CDNA5 async copy: global memory -> LDS, 16 bytes per lane, ASYNCcnt-tracked.
__device__ __forceinline__ void async_copy_b128(unsigned lds_byte_off,
                                                const void* gptr) {
  unsigned long long ga = (unsigned long long)gptr;
  asm volatile("global_load_async_to_lds_b128 %0, %1, off"
               :: "v"(lds_byte_off), "v"(ga) : "memory");
}

__device__ __forceinline__ void wait_async_zero() {
  asm volatile("s_wait_asynccnt 0x0" ::: "memory");
}

// ---------------------------------------------------------------------------
// f32 -> bf16 bulk convert (weights)
// ---------------------------------------------------------------------------
__global__ __launch_bounds__(256) void f32_to_bf16_v4(
    const float* __restrict__ in, __bf16* __restrict__ out, int n4) {
  int i = blockIdx.x * 256 + threadIdx.x;
  if (i < n4) {
    f32x4 v = ((const f32x4*)in)[i];
    bf16x4 o;
    o[0] = (__bf16)v[0]; o[1] = (__bf16)v[1];
    o[2] = (__bf16)v[2]; o[3] = (__bf16)v[3];
    ((bf16x4*)out)[i] = o;
  }
}

// ---------------------------------------------------------------------------
// Row RMSNorm (cols = 2048) -> bf16. One block (256 threads) per row.
// ---------------------------------------------------------------------------
__global__ __launch_bounds__(256) void rmsnorm_rows_bf16(
    const float* __restrict__ in, __bf16* __restrict__ out) {
  __shared__ float red[8];
  const int row = blockIdx.x;
  const int tid = threadIdx.x;
  const float* rp = in + (size_t)row * kDim + tid * 8;
  f32x4 a = *(const f32x4*)rp;
  f32x4 b = *(const f32x4*)(rp + 4);
  float s = 0.f;
#pragma unroll
  for (int i = 0; i < 4; ++i) s += a[i]*a[i] + b[i]*b[i];
#pragma unroll
  for (int d = 1; d < 32; d <<= 1) s += __shfl_xor(s, d, 32);
  if ((tid & 31) == 0) red[tid >> 5] = s;
  __syncthreads();
  float tot = 0.f;
#pragma unroll
  for (int w = 0; w < 8; ++w) tot += red[w];
  const float rs = rsqrtf(tot * (1.0f / kDim) + kEps);
  bf16x8 o;
#pragma unroll
  for (int i = 0; i < 4; ++i) { o[i] = (__bf16)(a[i]*rs); o[i+4] = (__bf16)(b[i]*rs); }
  *(bf16x8*)(out + (size_t)row * kDim + tid * 8) = o;
}

// ---------------------------------------------------------------------------
// GEMM: D[M,N] = A[M,K] * B[N,K]^T, bf16 inputs, f32 accumulate.
// Block tile 128x128x32, 256 threads = 8 waves, wave tile 32x64 (8 WMMA frags).
// Tiles staged via GLOBAL_LOAD_ASYNC_TO_LDS_B128, double-buffered, one
// s_barrier per K-step; loads for tile t+1 overlap WMMAs on tile t.
// EPI 0: f32 out; EPI 1: out = resid + acc*scale[col]; EPI 2: bf16(relu(acc)^2)
// ---------------------------------------------------------------------------
template<int EPI>
__global__ __launch_bounds__(256) void gemm_bf16_nt(
    const __bf16* __restrict__ A, const __bf16* __restrict__ B,
    int M, int N, int K,
    float* __restrict__ outf, __bf16* __restrict__ outb,
    const float* __restrict__ resid, const float* __restrict__ scale) {
  constexpr int BKP = 40;                       // 32 + 8 pad vs bank conflicts
  __shared__ __bf16 As[2][128][BKP];
  __shared__ __bf16 Bs[2][128][BKP];

  const int tid  = threadIdx.x;
  const int lane = tid & 31;
  const int wave = tid >> 5;
  const int wm   = wave >> 1;                   // 0..3 -> 32-row slice
  const int wn   = wave & 1;                    // 0..1 -> 64-col slice
  const int m0   = blockIdx.y * 128;
  const int n0   = blockIdx.x * 128;

  v8f acc[2][4];
#pragma unroll
  for (int mi = 0; mi < 2; ++mi)
#pragma unroll
    for (int nj = 0; nj < 4; ++nj) acc[mi][nj] = (v8f){};

  const int am = lane & 15;                     // A frag: row within 16
  const int ab = (lane >> 4) * 8;               // A frag: K base (ISA A layout)
  const int bn = lane & 15;                     // B frag: N within 16
  const int bb = (lane >> 4) * 8;               // B frag: K base (ISA layout)

  // This thread's two (row, col8) chunks of the 128x32 tile.
  const int r0 = tid >> 2,          c0 = (tid & 3) * 8;
  const int r1 = (tid + 256) >> 2,  c1 = ((tid + 256) & 3) * 8;

  auto issue_tile = [&](int k0, int buf) {
    async_copy_b128(lds_off(&As[buf][r0][c0]), &A[(size_t)(m0 + r0) * K + k0 + c0]);
    async_copy_b128(lds_off(&Bs[buf][r0][c0]), &B[(size_t)(n0 + r0) * K + k0 + c0]);
    async_copy_b128(lds_off(&As[buf][r1][c1]), &A[(size_t)(m0 + r1) * K + k0 + c1]);
    async_copy_b128(lds_off(&Bs[buf][r1][c1]), &B[(size_t)(n0 + r1) * K + k0 + c1]);
  };

  issue_tile(0, 0);
  const int nIter = K / 32;
  for (int it = 0; it < nIter; ++it) {
    const int cur = it & 1;
    wait_async_zero();        // my async loads for tile `it` have landed in LDS
    __syncthreads();          // everyone's loads landed; prev compute finished
    if (it + 1 < nIter) issue_tile((it + 1) * 32, cur ^ 1);

    v16bf af[2], bf[4];
#pragma unroll
    for (int mi = 0; mi < 2; ++mi) {
      const __bf16* p = &As[cur][wm * 32 + mi * 16 + am][ab];
      af[mi] = mk16(*(const bf16x8*)p, *(const bf16x8*)(p + 16));
    }
#pragma unroll
    for (int nj = 0; nj < 4; ++nj) {
      const __bf16* p = &Bs[cur][wn * 64 + nj * 16 + bn][bb];
      bf[nj] = mk16(*(const bf16x8*)p, *(const bf16x8*)(p + 16));
    }
#pragma unroll
    for (int mi = 0; mi < 2; ++mi)
#pragma unroll
      for (int nj = 0; nj < 4; ++nj)
        acc[mi][nj] = wmma_bf16(af[mi], bf[nj], acc[mi][nj]);
  }

  // Epilogue. C layout: elem i of v8f -> M = i + (lane>=16 ? 8:0), N = lane&15.
  const int mhi = (lane >> 4) * 8;
#pragma unroll
  for (int mi = 0; mi < 2; ++mi)
#pragma unroll
    for (int nj = 0; nj < 4; ++nj)
#pragma unroll
      for (int i = 0; i < 8; ++i) {
        int row = m0 + wm * 32 + mi * 16 + i + mhi;
        int col = n0 + wn * 64 + nj * 16 + (lane & 15);
        size_t idx = (size_t)row * N + col;
        float v = acc[mi][nj][i];
        if (EPI == 0) {
          outf[idx] = v;
        } else if (EPI == 1) {
          outf[idx] = resid[idx] + v * scale[col];
        } else {
          float r = v > 0.f ? v : 0.f;
          outb[idx] = (__bf16)(r * r);
        }
      }
}

// ---------------------------------------------------------------------------
// QK rmsnorm + rotary + layout shuffle. One 128-thread block per (n,t,h).
// Writes bf16 Q/K as [N,H,T,hd], bf16 V transposed as [N,H,hd,T],
// and f32 k,v into d_out sections.
// ---------------------------------------------------------------------------
__global__ __launch_bounds__(128) void qknorm_rotary(
    const float* __restrict__ qkv, __bf16* __restrict__ Qh,
    __bf16* __restrict__ Kh, __bf16* __restrict__ VT,
    float* __restrict__ out_base) {
  __shared__ float sq[kHd], sk[kHd];
  __shared__ float redq[4], redk[4];
  const int b = blockIdx.x;
  const int h = b % kNH;
  const int t = (b / kNH) % kT;
  const int n = b / (kNH * kT);
  const int d = threadIdx.x;

  const float* base = qkv + (size_t)(n * kT + t) * (3 * kDim) + h * kHd;
  const float qv = base[d];
  const float kv = base[kDim + d];
  const float vv = base[2 * kDim + d];

  float q2 = qv * qv, k2 = kv * kv;
#pragma unroll
  for (int s = 1; s < 32; s <<= 1) {
    q2 += __shfl_xor(q2, s, 32);
    k2 += __shfl_xor(k2, s, 32);
  }
  if ((d & 31) == 0) { redq[d >> 5] = q2; redk[d >> 5] = k2; }
  __syncthreads();
  const float sumq = redq[0] + redq[1] + redq[2] + redq[3];
  const float sumk = redk[0] + redk[1] + redk[2] + redk[3];
  const float rq = rsqrtf(sumq * (1.0f / kHd) + kEps);
  const float rk = rsqrtf(sumk * (1.0f / kHd) + kEps);
  sq[d] = qv * rq;
  sk[d] = kv * rk;
  __syncthreads();

  // freqs: (1/1e4)^(i/31) for i<32, else 0; pairs (d, d+64)
  const int idx = d & 63;
  const float f = (idx < 32) ? __expf((float)idx * (-9.210340371976184f / 31.0f)) : 0.0f;
  float sn, cs;
  __sincosf((float)t * f, &sn, &cs);
  float yq, yk;
  if (d < 64) {
    yq =  sq[d] * cs + sq[d + 64] * sn;
    yk =  sk[d] * cs + sk[d + 64] * sn;
  } else {
    yq = -sq[d - 64] * sn + sq[d] * cs;
    yk = -sk[d - 64] * sn + sk[d] * cs;
  }

  const size_t ho = ((size_t)(n * kNH + h) * kT + t) * kHd + d;
  Qh[ho] = (__bf16)yq;
  Kh[ho] = (__bf16)yk;
  VT[((size_t)(n * kNH + h) * kHd + d) * kT + t] = (__bf16)vv;

  const size_t ro = (size_t)(n * kT + t) * kDim + h * kHd + d;
  float* outk = out_base + (size_t)kRows * kDim;
  float* outv = outk + (size_t)kRows * kDim;
  outk[ro] = yk;
  outv[ro] = vv;
}

// ---------------------------------------------------------------------------
// Causal flash attention. One wave per (n, h, 16-query tile); 32 keys/iter.
// Q,K: [N,H,T,hd] bf16. VT: [N,H,hd,T] bf16. O: [N,T,DIM] bf16.
// ---------------------------------------------------------------------------
__global__ __launch_bounds__(32) void flash_attn(
    const __bf16* __restrict__ Q, const __bf16* __restrict__ Kb,
    const __bf16* __restrict__ VT, __bf16* __restrict__ O) {
  __shared__ __bf16 Pt[16][40];
  const int lane = threadIdx.x;
  constexpr int nqt = kT / 16;
  const int b  = blockIdx.x;
  const int qi = b % nqt;
  const int h  = (b / nqt) % kNH;
  const int n  = b / (nqt * kNH);
  const int q0 = qi * 16;

  const __bf16* qh = Q  + (size_t)(n * kNH + h) * kT * kHd;
  const __bf16* kh = Kb + (size_t)(n * kNH + h) * kT * kHd;
  const __bf16* vh = VT + (size_t)(n * kNH + h) * kHd * kT;

  const int am  = lane & 15;
  const int ab  = (lane >> 4) * 8;
  const int bnn = lane & 15;
  const int bb  = (lane >> 4) * 8;
  const int mhi = (lane >> 4) * 8;

  v16bf qf[4];
#pragma unroll
  for (int c = 0; c < 4; ++c) {
    const __bf16* p = qh + (size_t)(q0 + am) * kHd + c * 32 + ab;
    qf[c] = mk16(*(const bf16x8*)p, *(const bf16x8*)(p + 16));
  }

  v8f o[8];
  float mr[8], lr[8];
#pragma unroll
  for (int j = 0; j < 8; ++j) o[j] = (v8f){};
#pragma unroll
  for (int i = 0; i < 8; ++i) { mr[i] = -1e30f; lr[i] = 0.f; }

  const float scale = 0.08838834764831845f;  // 1/sqrt(128)

  for (int kbase = 0; kbase < q0 + 16; kbase += 32) {
    v8f s0 = (v8f){}, s1 = (v8f){};
#pragma unroll
    for (int c = 0; c < 4; ++c) {
      const __bf16* p0 = kh + (size_t)(kbase + bnn) * kHd + c * 32 + bb;
      s0 = wmma_bf16(qf[c], mk16(*(const bf16x8*)p0, *(const bf16x8*)(p0 + 16)), s0);
      const __bf16* p1 = kh + (size_t)(kbase + 16 + bnn) * kHd + c * 32 + bb;
      s1 = wmma_bf16(qf[c], mk16(*(const bf16x8*)p1, *(const bf16x8*)(p1 + 16)), s1);
    }

#pragma unroll
    for (int i = 0; i < 8; ++i) {
      const int qa  = q0 + i + mhi;
      const int k0i = kbase + (lane & 15);
      float x0 = (k0i      <= qa) ? s0[i] * scale : -1e30f;
      float x1 = (k0i + 16 <= qa) ? s1[i] * scale : -1e30f;
      float mx = fmaxf(x0, x1);
#pragma unroll
      for (int d = 1; d < 16; d <<= 1) mx = fmaxf(mx, __shfl_xor(mx, d, 32));
      const float mnew = fmaxf(mr[i], mx);
      const float corr = __expf(mr[i] - mnew);
      const float p0 = __expf(x0 - mnew);
      const float p1 = __expf(x1 - mnew);
      float rs = p0 + p1;
#pragma unroll
      for (int d = 1; d < 16; d <<= 1) rs += __shfl_xor(rs, d, 32);
      lr[i] = lr[i] * corr + rs;
      mr[i] = mnew;
#pragma unroll
      for (int j = 0; j < 8; ++j) o[j][i] *= corr;
      const int m = i + mhi;
      Pt[m][lane & 15]        = (__bf16)p0;
      Pt[m][16 + (lane & 15)] = (__bf16)p1;
    }
    __syncthreads();

    const __bf16* pp = &Pt[am][ab];
    v16bf pf = mk16(*(const bf16x8*)pp, *(const bf16x8*)(pp + 16));
#pragma unroll
    for (int j = 0; j < 8; ++j) {
      const __bf16* vp = vh + (size_t)(j * 16 + (lane & 15)) * kT + kbase + bb;
      o[j] = wmma_bf16(pf, mk16(*(const bf16x8*)vp, *(const bf16x8*)(vp + 16)), o[j]);
    }
    __syncthreads();
  }

#pragma unroll
  for (int i = 0; i < 8; ++i) {
    const int m = i + mhi;
    const float inv = 1.0f / lr[i];
    const size_t rowoff = ((size_t)(n * kT) + q0 + m) * kDim + h * kHd;
#pragma unroll
    for (int j = 0; j < 8; ++j)
      O[rowoff + j * 16 + (lane & 15)] = (__bf16)(o[j][i] * inv);
  }
}

// ---------------------------------------------------------------------------
// Host-side orchestration
// ---------------------------------------------------------------------------
extern "C" void kernel_launch(void* const* d_in, const int* in_sizes, int n_in,
                              void* d_out, int out_size, void* d_ws, size_t ws_size,
                              hipStream_t stream) {
  (void)in_sizes; (void)n_in; (void)out_size; (void)ws_size;
  const float* x         = (const float*)d_in[0];
  const float* qkv_w     = (const float*)d_in[1];
  const float* o_w       = (const float*)d_in[2];
  const float* o_scale   = (const float*)d_in[3];
  const float* mlp_w1    = (const float*)d_in[4];
  const float* mlp_w2    = (const float*)d_in[5];
  const float* mlp_scale = (const float*)d_in[6];
  float* out = (float*)d_out;   // [x | k | v], each kRows*kDim f32

  char* ws = (char*)d_ws;
  size_t off = 0;
  auto alloc = [&](size_t bytes) -> void* {
    void* p = ws + off;
    off += (bytes + 255) & ~(size_t)255;
    return p;
  };

  __bf16* qkvw_b = (__bf16*)alloc((size_t)3 * kDim * kDim * 2);
  __bf16* ow_b   = (__bf16*)alloc((size_t)kDim * kDim * 2);
  __bf16* w1_b   = (__bf16*)alloc((size_t)4 * kDim * kDim * 2);
  __bf16* w2_b   = (__bf16*)alloc((size_t)4 * kDim * kDim * 2);
  __bf16* xn_b   = (__bf16*)alloc((size_t)kRows * kDim * 2);     // also attn O
  float*  qkvf   = (float*) alloc((size_t)kRows * 3 * kDim * 4); // also x2 + h
  __bf16* qh_b   = (__bf16*)alloc((size_t)kRows * kDim * 2);     // also xn2
  __bf16* kh_b   = (__bf16*)alloc((size_t)kRows * kDim * 2);
  __bf16* vT_b   = (__bf16*)alloc((size_t)kRows * kDim * 2);

  float*  x2_f32 = qkvf;                                          // 64 MB
  __bf16* h_b    = (__bf16*)((char*)qkvf + (size_t)kRows * kDim * 4); // 128 MB
  __bf16* xn2_b  = qh_b;                                          // reuse after flash
  __bf16* attnO  = xn_b;                                          // reuse after GEMM1

  auto cvt = [&](const float* src, __bf16* dst, size_t n) {
    int n4 = (int)(n / 4);
    f32_to_bf16_v4<<<(n4 + 255) / 256, 256, 0, stream>>>(src, dst, n4);
  };
  cvt(qkv_w,  qkvw_b, (size_t)3 * kDim * kDim);
  cvt(o_w,    ow_b,   (size_t)kDim * kDim);
  cvt(mlp_w1, w1_b,   (size_t)4 * kDim * kDim);
  cvt(mlp_w2, w2_b,   (size_t)4 * kDim * kDim);

  // 1. xn = rmsnorm(x) (bf16)
  rmsnorm_rows_bf16<<<kRows, 256, 0, stream>>>(x, xn_b);
  // 2. qkv = xn @ qkv_w^T  (f32 out)
  gemm_bf16_nt<0><<<dim3((3 * kDim) / 128, kRows / 128), 256, 0, stream>>>(
      xn_b, qkvw_b, kRows, 3 * kDim, kDim, qkvf, nullptr, nullptr, nullptr);
  // 3. qk-norm + rotary + head-major layout; writes k/v f32 sections of d_out
  qknorm_rotary<<<kN * kT * kNH, 128, 0, stream>>>(qkvf, qh_b, kh_b, vT_b, out);
  // 4. causal flash attention -> attnO bf16 [N,T,DIM]
  flash_attn<<<kN * kNH * (kT / 16), 32, 0, stream>>>(qh_b, kh_b, vT_b, attnO);
  // 5. x2 = x + (attnO @ o_w^T) * o_scale
  gemm_bf16_nt<1><<<dim3(kDim / 128, kRows / 128), 256, 0, stream>>>(
      attnO, ow_b, kRows, kDim, kDim, x2_f32, nullptr, x, o_scale);
  // 6. xn2 = rmsnorm(x2)
  rmsnorm_rows_bf16<<<kRows, 256, 0, stream>>>(x2_f32, xn2_b);
  // 7. h = relu(xn2 @ w1^T)^2  (bf16)
  gemm_bf16_nt<2><<<dim3((4 * kDim) / 128, kRows / 128), 256, 0, stream>>>(
      xn2_b, w1_b, kRows, 4 * kDim, kDim, nullptr, h_b, nullptr, nullptr);
  // 8. out_x = x2 + (h @ w2^T) * mlp_scale
  gemm_bf16_nt<1><<<dim3(kDim / 128, kRows / 128), 256, 0, stream>>>(
      h_b, w2_b, kRows, kDim, 4 * kDim, out, nullptr, x2_f32, mlp_scale);
}
